// LagunaMoE_68264210203005
// MI455X (gfx1250) — matrix operand order repaired and checked by reading, not compiled, not run on hardware
//
#include <hip/hip_runtime.h>
#include <hip/hip_bf16.h>

#define HIDDEN        2048
#define NUM_EXPERTS   16
#define TOP_K         4
#define MOE_INTER     1024
#define SHARED_INTER  4096
#define NUM_TOKENS    2048
#define TOTAL_ROWS    (NUM_TOKENS * TOP_K)   // 8192 grouped-GEMM rows

typedef __attribute__((ext_vector_type(16))) __bf16 v16bf;
typedef __attribute__((ext_vector_type(8)))  float  v8f;

// D = A(16x32 bf16) * B(32x16 bf16) + C(16x16 f32), wave32 WMMA
#define WMMA_BF16(a, b, c) \
  __builtin_amdgcn_wmma_f32_16x16x32_bf16(false, (a), false, (b), (short)0, (c), false, false)

// ---------------------------------------------------------------------------
// 1) fp32 -> bf16 activation conversion
// ---------------------------------------------------------------------------
__global__ __launch_bounds__(256) void k_tobf16(const float* __restrict__ src,
                                                __bf16* __restrict__ dst, int n) {
  int i = blockIdx.x * 256 + threadIdx.x;
  if (i < n) dst[i] = (__bf16)src[i];
}

// ---------------------------------------------------------------------------
// 2) Router: logits = x @ gate_w^T, sigmoid, +bias for selection, top-4,
//    renormalized original scores. One block per token.
// ---------------------------------------------------------------------------
__global__ __launch_bounds__(256) void k_router(const float* __restrict__ x,
                                                const float* __restrict__ gate_w,
                                                const float* __restrict__ e_bias,
                                                int* __restrict__ top_idx,
                                                float* __restrict__ top_w) {
  const int t = blockIdx.x;
  const int tid = threadIdx.x;
  __shared__ float partial[256];
  __shared__ float logits[NUM_EXPERTS];

  const int e = tid >> 4;     // 16 threads per expert
  const int p = tid & 15;     // 128-element chunk per thread
  const float* xr = x + (size_t)t * HIDDEN + p * 128;
  const float* wr = gate_w + (size_t)e * HIDDEN + p * 128;
  float s = 0.f;
#pragma unroll 4
  for (int i = 0; i < 128; ++i) s += xr[i] * wr[i];
  partial[tid] = s;
  __syncthreads();
  if (p == 0) {
    float l = 0.f;
    for (int i = 0; i < 16; ++i) l += partial[(e << 4) + i];  // fixed order
    logits[e] = l;
  }
  __syncthreads();
  if (tid == 0) {
    float sc[NUM_EXPERTS], corr[NUM_EXPERTS];
    for (int i = 0; i < NUM_EXPERTS; ++i) {
      sc[i] = 1.f / (1.f + __expf(-logits[i]));
      corr[i] = sc[i] + e_bias[i];
    }
    int idx[TOP_K];
    float wsum = 0.f;
    for (int k = 0; k < TOP_K; ++k) {
      int best = -1;
      float bv = -1e30f;
      for (int i = 0; i < NUM_EXPERTS; ++i) {
        bool taken = false;
        for (int j = 0; j < k; ++j) taken |= (idx[j] == i);
        if (!taken && corr[i] > bv) { bv = corr[i]; best = i; }  // lowest idx wins ties
      }
      idx[k] = best;
      wsum += sc[best];
    }
    float inv = 1.f / wsum;
    for (int k = 0; k < TOP_K; ++k) {
      top_idx[t * TOP_K + k] = idx[k];
      top_w[t * TOP_K + k] = sc[idx[k]] * inv;
    }
  }
}

// ---------------------------------------------------------------------------
// 3) Expert counts -> exclusive offsets (integer atomics only: deterministic)
// ---------------------------------------------------------------------------
__global__ __launch_bounds__(256) void k_offsets(const int* __restrict__ top_idx,
                                                 int* __restrict__ offs /*[E+1]*/) {
  __shared__ int cnt[NUM_EXPERTS];
  int tid = threadIdx.x;
  if (tid < NUM_EXPERTS) cnt[tid] = 0;
  __syncthreads();
  for (int i = tid; i < NUM_TOKENS * TOP_K; i += 256) atomicAdd(&cnt[top_idx[i]], 1);
  __syncthreads();
  if (tid == 0) {
    int acc = 0;
    for (int e = 0; e < NUM_EXPERTS; ++e) { offs[e] = acc; acc += cnt[e]; }
    offs[NUM_EXPERTS] = acc;  // == TOTAL_ROWS
  }
}

// ---------------------------------------------------------------------------
// 4) Deterministic bucket build: one wave32 per expert, token-ordered append
//    via ballot + prefix-popcount. Records row_of[t][k] for the final gather.
// ---------------------------------------------------------------------------
__global__ __launch_bounds__(32) void k_buckets(const int* __restrict__ top_idx,
                                                const int* __restrict__ offs,
                                                int* __restrict__ bucket_tok,
                                                int* __restrict__ row_of) {
  const int e = blockIdx.x;
  const int lane = threadIdx.x;
  const int base = offs[e];
  int pos = 0;
  for (int t0 = 0; t0 < NUM_TOKENS; t0 += 32) {
    const int t = t0 + lane;
    int kf = -1;
#pragma unroll
    for (int k = 0; k < TOP_K; ++k)
      if (top_idx[t * TOP_K + k] == e) kf = k;  // top-k indices are distinct
    unsigned mask = (unsigned)__ballot(kf >= 0);  // wave32: low 32 bits
    if (kf >= 0) {
      int row = base + pos + __popc(mask & ((1u << lane) - 1u));
      bucket_tok[row] = t;
      row_of[t * TOP_K + kf] = row;
    }
    pos += __popc(mask);
  }
}

// ---------------------------------------------------------------------------
// 5) Fused gate+up GEMM + SwiGLU epilogue (bf16 WMMA).
//    Block = 256 thr (8 waves) computes a 16(M) x 128(N) tile of BOTH
//    g = A*Bg and u = A*Bu, then writes silu(g)*u as bf16.
//    A rows optionally gathered through rowmap (token ids per bucket row).
// ---------------------------------------------------------------------------
__global__ __launch_bounds__(256) void k_gateup(
    const __bf16* __restrict__ A, int lda,
    const int* __restrict__ rowmap,      // nullable: global row -> A row
    const int* __restrict__ offs,        // nullable: [E+1] segment offsets
    int Mtotal,                          // used when offs == nullptr
    const float* __restrict__ Bg, const float* __restrict__ Bu,
    long long strideBe, int ldb,
    __bf16* __restrict__ Out, int ldo, int K) {
  const int e = blockIdx.y;
  int base = 0, cnt = Mtotal;
  if (offs) { base = offs[e]; cnt = offs[e + 1] - base; }
  const int mt = blockIdx.z * 16;
  if (mt >= cnt) return;
  const int n0 = blockIdx.x * 128;

  __shared__ __bf16 sA[16][40];       // 16x32 A tile (+pad)
  __shared__ __bf16 sBg[32][136];     // 32x128 B tiles (+pad)
  __shared__ __bf16 sBu[32][136];
  __shared__ int sArow[16];

  const int tid = threadIdx.x;
  const int lane = tid & 31;
  const int wave = tid >> 5;
  const int hi = lane >> 4;   // lane half (K split for A, row offset for C/D)
  const int lr = lane & 15;   // A row / B,C,D column within 16-wide tile

  if (tid < 16) {
    int r = mt + tid;
    int rr = (r < cnt) ? r : (cnt - 1);   // clamp partial tile (dup last row)
    sArow[tid] = rowmap ? rowmap[base + rr] : (base + rr);
  }
  __syncthreads();

  const float* bg = Bg + (long long)e * strideBe;
  const float* bu = Bu + (long long)e * strideBe;

  v8f cg = {};
  v8f cu = {};

  for (int k0 = 0; k0 < K; k0 += 32) {
    // ---- stage A (16 rows x 32 bf16): 128 threads x 8B ----
    if (tid < 128) {
      int row = tid >> 3, seg = tid & 7;
      const __bf16* src = A + (size_t)sArow[row] * lda + k0 + seg * 4;
      *(uint2*)&sA[row][seg * 4] = *(const uint2*)src;
    }
    // ---- stage Bg/Bu (32 x 128 fp32 -> bf16 in LDS), coalesced float4 ----
#pragma unroll
    for (int i = tid; i < 1024; i += 256) {
      int el = i << 2;
      int kr = el >> 7, c = el & 127;
      size_t go = (size_t)(k0 + kr) * ldb + n0 + c;
      float4 vg = *(const float4*)(bg + go);
      float4 vu = *(const float4*)(bu + go);
      sBg[kr][c + 0] = (__bf16)vg.x; sBg[kr][c + 1] = (__bf16)vg.y;
      sBg[kr][c + 2] = (__bf16)vg.z; sBg[kr][c + 3] = (__bf16)vg.w;
      sBu[kr][c + 0] = (__bf16)vu.x; sBu[kr][c + 1] = (__bf16)vu.y;
      sBu[kr][c + 2] = (__bf16)vu.z; sBu[kr][c + 3] = (__bf16)vu.w;
    }
    if (k0 + 32 < K)  // hint next B tile toward L2 (global_prefetch_b8)
      __builtin_prefetch(bg + (size_t)(k0 + 32 + (tid >> 3)) * ldb + n0 + ((tid & 7) << 4), 0, 3);
    __syncthreads();

    // ---- build fragments in contiguous k order (k = hi*16 + j, A and B alike) ----
    v16bf a, fg, fu;
#pragma unroll
    for (int j = 0; j < 8; ++j) {
      a[j]     = sA[lr][hi * 16 + j];          // K = hi*16 + j
      a[j + 8] = sA[lr][hi * 16 + 8 + j];      // K = hi*16 + 8 + j
    }
    const int c = (wave << 4) + lr;            // column within 128-wide tile
#pragma unroll
    for (int j = 0; j < 16; ++j) {
      fg[j] = sBg[hi * 16 + j][c];             // K = hi*16 + j
      fu[j] = sBu[hi * 16 + j][c];
    }
    cg = WMMA_BF16(a, fg, cg);
    cu = WMMA_BF16(a, fu, cu);
    __syncthreads();
  }

  // ---- SwiGLU epilogue: Out = silu(g) * u (bf16) ----
  const int outc = n0 + (wave << 4) + lr;
#pragma unroll
  for (int r = 0; r < 8; ++r) {
    int rl = mt + (hi << 3) + r;               // C/D row = r + 8*hi
    if (rl < cnt) {
      float g = cg[r], u = cu[r];
      float sg = g / (1.f + __expf(-g));
      Out[(size_t)(base + rl) * ldo + outc] = (__bf16)(sg * u);
    }
  }
}

// ---------------------------------------------------------------------------
// 6) Down-projection GEMM (bf16 WMMA, fp32 output). Rows are already the
//    compacted bucket rows (or plain token rows for the shared expert).
// ---------------------------------------------------------------------------
__global__ __launch_bounds__(256) void k_down(
    const __bf16* __restrict__ A, int lda,
    const int* __restrict__ offs, int Mtotal,
    const float* __restrict__ Bw, long long strideBe, int ldb,
    float* __restrict__ Out, int ldo, int K) {
  const int e = blockIdx.y;
  int base = 0, cnt = Mtotal;
  if (offs) { base = offs[e]; cnt = offs[e + 1] - base; }
  const int mt = blockIdx.z * 16;
  if (mt >= cnt) return;
  const int n0 = blockIdx.x * 128;

  __shared__ __bf16 sA[16][40];
  __shared__ __bf16 sB[32][136];
  __shared__ int sArow[16];

  const int tid = threadIdx.x;
  const int lane = tid & 31;
  const int wave = tid >> 5;
  const int hi = lane >> 4;
  const int lr = lane & 15;

  if (tid < 16) {
    int r = mt + tid;
    int rr = (r < cnt) ? r : (cnt - 1);
    sArow[tid] = base + rr;
  }
  __syncthreads();

  const float* bw = Bw + (long long)e * strideBe;
  v8f acc = {};

  for (int k0 = 0; k0 < K; k0 += 32) {
    if (tid < 128) {
      int row = tid >> 3, seg = tid & 7;
      const __bf16* src = A + (size_t)sArow[row] * lda + k0 + seg * 4;
      *(uint2*)&sA[row][seg * 4] = *(const uint2*)src;
    }
#pragma unroll
    for (int i = tid; i < 1024; i += 256) {
      int el = i << 2;
      int kr = el >> 7, c = el & 127;
      float4 v = *(const float4*)(bw + (size_t)(k0 + kr) * ldb + n0 + c);
      sB[kr][c + 0] = (__bf16)v.x; sB[kr][c + 1] = (__bf16)v.y;
      sB[kr][c + 2] = (__bf16)v.z; sB[kr][c + 3] = (__bf16)v.w;
    }
    if (k0 + 32 < K)
      __builtin_prefetch(bw + (size_t)(k0 + 32 + (tid >> 3)) * ldb + n0 + ((tid & 7) << 4), 0, 3);
    __syncthreads();

    v16bf a, fb;
#pragma unroll
    for (int j = 0; j < 8; ++j) {
      a[j]     = sA[lr][hi * 16 + j];
      a[j + 8] = sA[lr][hi * 16 + 8 + j];
    }
    const int c = (wave << 4) + lr;
#pragma unroll
    for (int j = 0; j < 16; ++j) fb[j] = sB[hi * 16 + j][c];
    acc = WMMA_BF16(a, fb, acc);
    __syncthreads();
  }

  const int outc = n0 + (wave << 4) + lr;
#pragma unroll
  for (int r = 0; r < 8; ++r) {
    int rl = mt + (hi << 3) + r;
    if (rl < cnt) Out[(size_t)(base + rl) * ldo + outc] = acc[r];
  }
}

// ---------------------------------------------------------------------------
// 7) Final combine: out[t] = sum_k top_w[t][k] * Y[row_of[t][k]] + shared[t]
//    Fixed k-order summation -> bitwise deterministic.
// ---------------------------------------------------------------------------
__global__ __launch_bounds__(256) void k_combine(const float* __restrict__ Y,
                                                 const float* __restrict__ SD,
                                                 const int* __restrict__ row_of,
                                                 const float* __restrict__ top_w,
                                                 float* __restrict__ out) {
  const int t = blockIdx.x;
  const int h = blockIdx.y * 256 + threadIdx.x;
  float acc = SD[(size_t)t * HIDDEN + h];
#pragma unroll
  for (int k = 0; k < TOP_K; ++k)
    acc += top_w[t * TOP_K + k] * Y[(size_t)row_of[t * TOP_K + k] * HIDDEN + h];
  out[(size_t)t * HIDDEN + h] = acc;
}

// ---------------------------------------------------------------------------
// Launch
// ---------------------------------------------------------------------------
extern "C" void kernel_launch(void* const* d_in, const int* in_sizes, int n_in,
                              void* d_out, int out_size, void* d_ws, size_t ws_size,
                              hipStream_t stream) {
  const float* x       = (const float*)d_in[0];
  const float* gate_w  = (const float*)d_in[1];
  const float* e_bias  = (const float*)d_in[2];
  const float* w_gate  = (const float*)d_in[3];
  const float* w_up    = (const float*)d_in[4];
  const float* w_down  = (const float*)d_in[5];
  const float* sw_gate = (const float*)d_in[6];
  const float* sw_up   = (const float*)d_in[7];
  const float* sw_down = (const float*)d_in[8];
  float* out = (float*)d_out;

  char* ws = (char*)d_ws;
  size_t off = 0;
  auto alloc = [&](size_t bytes) -> void* {
    void* p = (void*)(ws + off);
    off += (bytes + 255) & ~(size_t)255;
    return p;
  };
  __bf16* xb     = (__bf16*)alloc((size_t)NUM_TOKENS * HIDDEN * 2);
  int*    topidx = (int*)  alloc((size_t)NUM_TOKENS * TOP_K * 4);
  float*  topw   = (float*)alloc((size_t)NUM_TOKENS * TOP_K * 4);
  int*    offs   = (int*)  alloc((NUM_EXPERTS + 1) * 4);
  int*    btok   = (int*)  alloc((size_t)TOTAL_ROWS * 4);
  int*    rowof  = (int*)  alloc((size_t)NUM_TOKENS * TOP_K * 4);
  __bf16* Abuf   = (__bf16*)alloc((size_t)TOTAL_ROWS * MOE_INTER * 2);
  float*  Y      = (float*) alloc((size_t)TOTAL_ROWS * HIDDEN * 4);
  __bf16* S      = (__bf16*)alloc((size_t)NUM_TOKENS * SHARED_INTER * 2);
  float*  SD     = (float*) alloc((size_t)NUM_TOKENS * HIDDEN * 4);

  // 1) activations -> bf16
  k_tobf16<<<(NUM_TOKENS * HIDDEN) / 256, 256, 0, stream>>>(x, xb, NUM_TOKENS * HIDDEN);
  // 2) routing
  k_router<<<NUM_TOKENS, 256, 0, stream>>>(x, gate_w, e_bias, topidx, topw);
  // 3) offsets, 4) buckets
  k_offsets<<<1, 256, 0, stream>>>(topidx, offs);
  k_buckets<<<NUM_EXPERTS, 32, 0, stream>>>(topidx, offs, btok, rowof);

  // 5) expert gate/up (grouped): grid = (N/128, E, maxMtiles)
  {
    dim3 g(MOE_INTER / 128, NUM_EXPERTS, NUM_TOKENS / 16);
    k_gateup<<<g, 256, 0, stream>>>(xb, HIDDEN, btok, offs, 0,
                                    w_gate, w_up,
                                    (long long)HIDDEN * MOE_INTER, MOE_INTER,
                                    Abuf, MOE_INTER, HIDDEN);
  }
  // 6) expert down
  {
    dim3 g(HIDDEN / 128, NUM_EXPERTS, NUM_TOKENS / 16);
    k_down<<<g, 256, 0, stream>>>(Abuf, MOE_INTER, offs, 0,
                                  w_down, (long long)MOE_INTER * HIDDEN, HIDDEN,
                                  Y, HIDDEN, MOE_INTER);
  }
  // 7) shared gate/up
  {
    dim3 g(SHARED_INTER / 128, 1, NUM_TOKENS / 16);
    k_gateup<<<g, 256, 0, stream>>>(xb, HIDDEN, nullptr, nullptr, NUM_TOKENS,
                                    sw_gate, sw_up, 0, SHARED_INTER,
                                    S, SHARED_INTER, HIDDEN);
  }
  // 8) shared down
  {
    dim3 g(HIDDEN / 128, 1, NUM_TOKENS / 16);
    k_down<<<g, 256, 0, stream>>>(S, SHARED_INTER, nullptr, NUM_TOKENS,
                                  sw_down, 0, HIDDEN,
                                  SD, HIDDEN, SHARED_INTER);
  }
  // 9) combine
  {
    dim3 g(NUM_TOKENS, HIDDEN / 256);
    k_combine<<<g, 256, 0, stream>>>(Y, SD, rowof, topw, out);
  }
}
